// Resonator_82471962017879
// MI455X (gfx1250) — 8-bit matrix operand order repaired and checked by reading, compiled, not run on hardware
//
#include <hip/hip_runtime.h>
#include <stdint.h>


typedef int i32x8 __attribute__((ext_vector_type(8)));

#define BB 64
#define FF 4
#define VV 256
#define DD 16384
#define ITERS 10


__global__ void pack_sign_s8(const float* __restrict__ src, int8_t* __restrict__ dst, int n) {
    int i = blockIdx.x * blockDim.x + threadIdx.x;
    if (i < n) dst[i] = (src[i] >= 0.0f) ? (int8_t)1 : (int8_t)-1;
}

__global__ void pack_codebooks(const float* __restrict__ src,
                               int8_t* __restrict__ cb8, int8_t* __restrict__ cbT8) {
    int i = blockIdx.x * blockDim.x + threadIdx.x;
    int f = i / (VV * DD);
    int r = i - f * (VV * DD);
    int v = r / DD;
    int d = r - v * DD;
    int8_t s = (src[i] >= 0.0f) ? (int8_t)1 : (int8_t)-1;
    cb8[i] = s;
    cbT8[((size_t)f * DD + d) * VV + v] = s;
}

__global__ void phaseA(const int8_t* __restrict__ in8,
                       const int8_t* __restrict__ est8,
                       int8_t* __restrict__ ne8) {
    int idx = blockIdx.x * blockDim.x + threadIdx.x;
    int b = idx >> 14;
    int d = idx & (DD - 1);
    const int8_t* e = est8 + (size_t)b * FF * DD + d;
    int8_t e0 = e[0], e1 = e[DD], e2 = e[2 * DD], e3 = e[3 * DD];
    int8_t p  = (int8_t)(e0 * e1 * e2 * e3);
    int8_t ip = (int8_t)(in8[(size_t)b * DD + d] * p);
    int8_t* o = ne8 + (size_t)b * FF * DD + d;
    o[0]      = (int8_t)(ip * e0);
    o[DD]     = (int8_t)(ip * e1);
    o[2 * DD] = (int8_t)(ip * e2);
    o[3 * DD] = (int8_t)(ip * e3);
}

__device__ __forceinline__ i32x8 load_a_frag(const int8_t* ar, int k) {
    int2 a01 = *(const int2*)(ar + k);
    int2 a23 = *(const int2*)(ar + k + 16);
    int2 a45 = *(const int2*)(ar + k + 32);
    int2 a67 = *(const int2*)(ar + k + 48);
    i32x8 a;
    a[0] = a01.x; a[1] = a01.y; a[2] = a23.x; a[3] = a23.y;
    a[4] = a45.x; a[5] = a45.y; a[6] = a67.x; a[7] = a67.y;
    return a;
}
__device__ __forceinline__ i32x8 load_b_frag(const int8_t* br, int k) {
    int2 b01 = *(const int2*)(br + k);
    int2 b23 = *(const int2*)(br + k + 16);
    int2 b45 = *(const int2*)(br + k + 32);
    int2 b67 = *(const int2*)(br + k + 48);
    i32x8 b;
    b[0] = b01.x; b[1] = b01.y; b[2] = b23.x; b[3] = b23.y;
    b[4] = b45.x; b[5] = b45.y; b[6] = b67.x; b[7] = b67.y;
    return b;
}

__device__ __forceinline__ i32x8 simtile_iu8(const int8_t* __restrict__ A,
                                             const int8_t* __restrict__ cb8,
                                             int f, int m0, int n0, int lane) {
    int row = lane & 15, hi = lane >> 4;
    const int8_t* ar = A   + ((size_t)(m0 + row) * FF + f) * DD + 8 * hi;
    const int8_t* br = cb8 + ((size_t)f * VV + (n0 + row)) * DD + 8 * hi;
    i32x8 accA = {}, accB = {};
#pragma unroll 1
    for (int k = 0; k < DD; k += 128) {
        __builtin_prefetch(br + k + 512, 0, 1);
        i32x8 a0 = load_a_frag(ar, k);
        i32x8 b0 = load_b_frag(br, k);
        i32x8 a1 = load_a_frag(ar, k + 64);
        i32x8 b1 = load_b_frag(br, k + 64);
        accA = __builtin_amdgcn_wmma_i32_16x16x64_iu8(true, a0, true, b0, accA, false, false);
        accB = __builtin_amdgcn_wmma_i32_16x16x64_iu8(true, a1, true, b1, accB, false, false);
    }
    return accA + accB;
}

__global__ void phaseB_planes(const int8_t* __restrict__ A, const int8_t* __restrict__ cb8,
                              uint8_t* __restrict__ sim0, int8_t* __restrict__ sim1) {
    int wave = (blockIdx.x * blockDim.x + threadIdx.x) >> 5;
    int lane = threadIdx.x & 31;
    int f  = wave >> 6;
    int m0 = ((wave >> 4) & 3) * 16;
    int n0 = (wave & 15) * 16;
    i32x8 acc = simtile_iu8(A, cb8, f, m0, n0, lane);
    int hi = lane >> 4;
    int vv = n0 + (lane & 15);
#pragma unroll
    for (int r = 0; r < 8; ++r) {
        size_t o = ((size_t)(m0 + r + 8 * hi) * FF + f) * VV + vv;
        int s = acc[r];
        sim1[o] = (int8_t)(s >> 8);
        sim0[o] = (uint8_t)(s & 255);
    }
}

__global__ void phaseB_sim32(const int8_t* __restrict__ A, const int8_t* __restrict__ cb8,
                             int* __restrict__ sim32) {
    int wave = (blockIdx.x * blockDim.x + threadIdx.x) >> 5;
    int lane = threadIdx.x & 31;
    int f  = wave >> 6;
    int m0 = ((wave >> 4) & 3) * 16;
    int n0 = (wave & 15) * 16;
    i32x8 acc = simtile_iu8(A, cb8, f, m0, n0, lane);
    int hi = lane >> 4;
    int vv = n0 + (lane & 15);
#pragma unroll
    for (int r = 0; r < 8; ++r)
        sim32[((size_t)(m0 + r + 8 * hi) * FF + f) * VV + vv] = acc[r];
}

__global__ void phaseC(const uint8_t* __restrict__ sim0, const int8_t* __restrict__ sim1,
                       const int8_t* __restrict__ cbT8, int8_t* __restrict__ est8) {
    int wave = (blockIdx.x * blockDim.x + threadIdx.x) >> 5;
    int lane = threadIdx.x & 31;
    int f  = wave >> 10;
    int nt = wave & 1023;
    int n0 = nt * 16;
    int row = lane & 15, hi = lane >> 4;

    const int8_t* br = cbT8 + ((size_t)f * DD + (n0 + row)) * VV + 8 * hi;

    i32x8 acc0[4] = {}, acc1[4] = {};
#pragma unroll
    for (int k = 0; k < VV; k += 64) {
        i32x8 b = load_b_frag(br, k);
#pragma unroll
        for (int mt = 0; mt < 4; ++mt) {
            size_t arow = ((size_t)(mt * 16 + row) * FF + f) * VV + 8 * hi;
            i32x8 a1 = load_a_frag((const int8_t*)(sim1 + arow), k);
            i32x8 a0 = load_a_frag((const int8_t*)(sim0 + arow), k);
            acc1[mt] = __builtin_amdgcn_wmma_i32_16x16x64_iu8(true,  a1, true, b, acc1[mt], false, false);
            acc0[mt] = __builtin_amdgcn_wmma_i32_16x16x64_iu8(false, a0, true, b, acc0[mt], false, false);
        }
    }
    int dd = n0 + (lane & 15);
#pragma unroll
    for (int mt = 0; mt < 4; ++mt) {
#pragma unroll
        for (int r = 0; r < 8; ++r) {
            int val = (acc1[mt][r] << 8) + acc0[mt][r];
            int bb = mt * 16 + r + 8 * hi;
            est8[((size_t)bb * FF + f) * DD + dd] = (val >= 0) ? (int8_t)1 : (int8_t)-1;
        }
    }
}


__global__ void argmaxK(const int* __restrict__ sim32, float* __restrict__ out) {
    int t = blockIdx.x * blockDim.x + threadIdx.x;
    if (t >= BB * FF) return;
    const int* s = sim32 + (size_t)t * VV;
    int best = -1, bi = 0;
    for (int v = 0; v < VV; ++v) {
        int a = s[v];
        a = (a < 0) ? -a : a;
        if (a > best) { best = a; bi = v; }
    }
    out[t] = (float)bi;
}

__global__ void est_to_float(const int8_t* __restrict__ est8, float* __restrict__ out, int n) {
    int i = blockIdx.x * blockDim.x + threadIdx.x;
    if (i < n) out[i] = (float)est8[i];
}


extern "C" void kernel_launch(void* const* d_in, const int* in_sizes, int n_in,
                              void* d_out, int out_size, void* d_ws, size_t ws_size,
                              hipStream_t stream) {
    const float* f_input = (const float*)d_in[0];
    const float* f_est   = (const float*)d_in[1];
    const float* f_cb    = (const float*)d_in[2];

    char* ws = (char*)d_ws;
    size_t off = 0;
    auto carve = [&](size_t bytes) { char* p = ws + off; off = (off + bytes + 255) & ~(size_t)255; return p; };
    int8_t*  in8   = (int8_t*)carve((size_t)BB * DD);
    int8_t*  est8  = (int8_t*)carve((size_t)BB * FF * DD);
    int8_t*  ne8   = (int8_t*)carve((size_t)BB * FF * DD);
    int8_t*  cb8   = (int8_t*)carve((size_t)FF * VV * DD);
    int8_t*  cbT8  = (int8_t*)carve((size_t)FF * DD * VV);
    uint8_t* sim0  = (uint8_t*)carve((size_t)BB * FF * VV);
    int8_t*  sim1  = (int8_t*)carve((size_t)BB * FF * VV);
    int*     sim32 = (int*)carve((size_t)BB * FF * VV * 4);
    (void)ws_size; (void)in_sizes; (void)n_in; (void)out_size;

    float* out_est = (float*)d_out;
    float* out_idx = out_est + (size_t)BB * FF * DD;

    pack_sign_s8<<<(BB * DD) / 256, 256, 0, stream>>>(f_input, in8, BB * DD);
    pack_sign_s8<<<(BB * FF * DD) / 256, 256, 0, stream>>>(f_est, est8, BB * FF * DD);
    pack_codebooks<<<(FF * VV * DD) / 256, 256, 0, stream>>>(f_cb, cb8, cbT8);

    for (int it = 0; it < ITERS; ++it) {
        phaseA<<<(BB * DD) / 256, 256, 0, stream>>>(in8, est8, ne8);
        phaseB_planes<<<32, 256, 0, stream>>>(ne8, cb8, sim0, sim1);
        phaseC<<<512, 256, 0, stream>>>(sim0, sim1, cbT8, est8);
    }

    phaseB_sim32<<<32, 256, 0, stream>>>(est8, cb8, sim32);
    argmaxK<<<1, 256, 0, stream>>>(sim32, out_idx);
    est_to_float<<<(BB * FF * DD) / 256, 256, 0, stream>>>(est8, out_est, BB * FF * DD);
}
